// Block_3186865734349
// MI455X (gfx1250) — matrix operand order repaired and checked by reading, not compiled, not run on hardware
//
#include <hip/hip_runtime.h>

typedef __bf16 bf16;
typedef __attribute__((ext_vector_type(16))) __bf16 v16bf;
typedef __attribute__((ext_vector_type(8)))  __bf16 bf16x8;
typedef __attribute__((ext_vector_type(8)))  float  v8f;

static constexpr int BATCH = 2, SEQ = 2048, HEADS = 16, DHEAD = 64;
static constexpr int DM = 1024, DPROJ = 7168, DBI = 2048, DCAT = 3072;
static constexpr int TOKENS = BATCH * SEQ; // 4096

union Frag16 { v16bf v; bf16x8 h[2]; unsigned u[8]; };

// ---------------------------------------------------------------------------
// Pack f32 weights [K,N] into bf16 WMMA B-fragment order:
// element i -> e = i&15, lane = (i>>4)&31, tile = i>>9, nt = tile % (N/16),
// kt = tile / (N/16);  k = kt*32 + 16*(e>>3) + 8*(lane>>4) + (e&7) ; n = nt*16 + (lane&15).
// A wave then loads its 32x16 B-fragment as 32 contiguous bytes per lane.
// ---------------------------------------------------------------------------
__global__ void __launch_bounds__(256) pack_weights(const float* __restrict__ w,
                                                    bf16* __restrict__ wpk,
                                                    int K, int N) {
  const int nt16 = N >> 4;
  const int total = K * N;
  int i = blockIdx.x * 256 + threadIdx.x;
  const int stride = gridDim.x * 256;
  for (; i < total; i += stride) {
    const int e = i & 15;
    const int l = (i >> 4) & 31;
    const int tile = i >> 9;
    const int nt = tile % nt16;
    const int kt = tile / nt16;
    const int k = kt * 32 + ((e >> 3) << 4) + ((l >> 4) << 3) + (e & 7);
    const int n = nt * 16 + (l & 15);
    wpk[i] = (bf16)w[(size_t)k * N + n];
  }
}

// ---------------------------------------------------------------------------
// Input LayerNorm + token shift (scatter), f32 -> bf16
// channels [0,256): dest l+1 ; [256,896): dest l ; [896,1024): dest l+2
// ---------------------------------------------------------------------------
__global__ void __launch_bounds__(256) ln_shift_kernel(const float* __restrict__ x,
                                                       const float* __restrict__ g,
                                                       const float* __restrict__ be,
                                                       bf16* __restrict__ x16) {
  __shared__ float rs[256], rq[256];
  const int row = blockIdx.x;
  const int b = row / SEQ, l = row % SEQ;
  const int tid = threadIdx.x;
  const float* xr = x + (size_t)row * DM;
  float v[4], s = 0.f, q = 0.f;
#pragma unroll
  for (int i = 0; i < 4; ++i) {
    const int c = i * 256 + tid;
    v[i] = xr[c]; s += v[i]; q += v[i] * v[i];
  }
  rs[tid] = s; rq[tid] = q; __syncthreads();
  for (int off = 128; off; off >>= 1) {
    if (tid < off) { rs[tid] += rs[tid + off]; rq[tid] += rq[tid + off]; }
    __syncthreads();
  }
  const float mean = rs[0] * (1.f / DM);
  const float var  = rq[0] * (1.f / DM) - mean * mean;
  const float rstd = rsqrtf(var + 1e-5f);
#pragma unroll
  for (int i = 0; i < 4; ++i) {
    const int c = i * 256 + tid;
    const float y = (v[i] - mean) * rstd * g[c] + be[c];
    const int dl = (c < 256) ? (l + 1) : ((c < 896) ? l : (l + 2));
    if (dl < SEQ) x16[((size_t)b * SEQ + dl) * DM + c] = (bf16)y;
    if (l == 0 && (c < 256 || c >= 896)) x16[(size_t)b * SEQ * DM + c] = (bf16)0.f;
    if (l == 1 && c >= 896) x16[((size_t)b * SEQ + 1) * DM + c] = (bf16)0.f;
  }
}

// ---------------------------------------------------------------------------
// LDS-free bf16 WMMA GEMM: C[M,N] = A[M,K] * B[K,N].
// A row-major bf16 (fragment halves are contiguous 16B spans -> b128 loads).
// B pre-packed in fragment order (32B contiguous per lane -> b128 loads).
// One wave computes a 32(M) x 64(N) tile: 8 accumulators, 8 WMMA per K-step.
// No LDS, no barriers; reuse across waves is served by L2 (192 MB).
// ---------------------------------------------------------------------------
template <bool STORE_BF16>
__global__ void __launch_bounds__(256) gemm_wmma(const bf16* __restrict__ A,
                                                 const bf16* __restrict__ Bpk,
                                                 void* __restrict__ Cout,
                                                 int M, int N, int K) {
  const int lane = threadIdx.x & 31, ln = lane & 15, hf = lane >> 4;
  const int wave = threadIdx.x >> 5;
  const int wglob = blockIdx.x * 8 + wave;
  const int ntiles = N >> 6;
  const int wm = wglob / ntiles;
  const int wn = wglob % ntiles;
  const int mbase = wm * 32;
  const int nbase = wn * 64;
  const int nt16 = N >> 4;

  v8f acc[2][4] = {};

  const bf16* arow0 = A + (size_t)(mbase + ln) * K + 8 * hf;
  const bf16* arow1 = arow0 + (size_t)16 * K;
  const bf16* bbase = Bpk + ((size_t)(nbase >> 4) * 32 + lane) * 16;

  for (int kk = 0; kk < K; kk += 32) {
    Frag16 a0, a1;
    a0.h[0] = *(const bf16x8*)(arow0 + kk);
    a0.h[1] = *(const bf16x8*)(arow0 + kk + 16);
    a1.h[0] = *(const bf16x8*)(arow1 + kk);
    a1.h[1] = *(const bf16x8*)(arow1 + kk + 16);
    if (kk + 32 < K) {
      __builtin_prefetch(arow0 + kk + 32, 0, 1);
      __builtin_prefetch(arow1 + kk + 32, 0, 1);
    }
    const bf16* bp = bbase + (size_t)(kk >> 5) * nt16 * 32 * 16;
#pragma unroll
    for (int f = 0; f < 4; ++f) {
      Frag16 bfr;
      bfr.h[0] = ((const bf16x8*)(bp + (size_t)f * 32 * 16))[0];
      bfr.h[1] = ((const bf16x8*)(bp + (size_t)f * 32 * 16))[1];
      acc[0][f] = __builtin_amdgcn_wmma_f32_16x16x32_bf16(
          false, a0.v, false, bfr.v, (short)0, acc[0][f], false, false);
      acc[1][f] = __builtin_amdgcn_wmma_f32_16x16x32_bf16(
          false, a1.v, false, bfr.v, (short)0, acc[1][f], false, false);
    }
  }

  // C layout: lane = N col (ln), vgpr r = M row r (+8 for lanes 16-31)
#pragma unroll
  for (int s = 0; s < 2; ++s)
#pragma unroll
    for (int f = 0; f < 4; ++f) {
      const int n = nbase + f * 16 + ln;
#pragma unroll
      for (int r = 0; r < 8; ++r) {
        const int m = mbase + s * 16 + r + hf * 8;
        const float val = acc[s][f][r];
        if constexpr (STORE_BF16)
          ((bf16*)Cout)[(size_t)m * N + n] = (bf16)val;
        else
          ((float*)Cout)[(size_t)m * N + n] = val;
      }
    }
}

// ---------------------------------------------------------------------------
// Mid LayerNorm over 7168 + split:
//   q,k -> [b,h,l,d] bf16 ; v -> TRANSPOSED vt [b,h,d,l] bf16 (so attention's
//   V^T fragments are contiguous b128 loads) ; p1*p2 -> op[:,1024:3072].
// The row is staged into LDS with CDNA5 async-to-LDS loads (ASYNCcnt).
// ---------------------------------------------------------------------------
__global__ void __launch_bounds__(256) midln_split(const bf16* __restrict__ h16,
                                                   const float* __restrict__ g,
                                                   const float* __restrict__ be,
                                                   bf16* __restrict__ qb,
                                                   bf16* __restrict__ kb,
                                                   bf16* __restrict__ vt,
                                                   bf16* __restrict__ op) {
  __shared__ bf16 hrow[DPROJ];
  __shared__ float rs[256], rq[256];
  const int row = blockIdx.x;  // token index b*SEQ + l
  const int b = row / SEQ, l = row % SEQ;
  const int tid = threadIdx.x;
  const bf16* hr = h16 + (size_t)row * DPROJ;

  // Async stage: 896 x b128 chunks, tracked by ASYNCcnt.
  for (int i = tid; i < DPROJ / 8; i += 256) {
    const bf16* gsrc = hr + (size_t)i * 8;
    const unsigned loff = (unsigned)(size_t)&hrow[i * 8];
    asm volatile("global_load_async_to_lds_b128 %0, %1, off"
                 :: "v"(loff), "v"(gsrc)
                 : "memory");
  }
  asm volatile("s_wait_asynccnt 0" ::: "memory");
  __syncthreads();

  float s = 0.f, q = 0.f;
  for (int c = tid; c < DPROJ; c += 256) {
    const float v = (float)hrow[c];
    s += v; q += v * v;
  }
  rs[tid] = s; rq[tid] = q; __syncthreads();
  for (int off = 128; off; off >>= 1) {
    if (tid < off) { rs[tid] += rs[tid + off]; rq[tid] += rq[tid + off]; }
    __syncthreads();
  }
  const float mean = rs[0] * (1.f / DPROJ);
  const float var  = rq[0] * (1.f / DPROJ) - mean * mean;
  const float rstd = rsqrtf(var + 1e-5f);

  for (int c = tid; c < 3 * DM; c += 256) {
    const float y = ((float)hrow[c] - mean) * rstd * g[c] + be[c];
    if (c < DM) {
      qb[((size_t)(b * HEADS + (c >> 6)) * SEQ + l) * DHEAD + (c & 63)] = (bf16)y;
    } else if (c < 2 * DM) {
      const int cc = c - DM;
      kb[((size_t)(b * HEADS + (cc >> 6)) * SEQ + l) * DHEAD + (cc & 63)] = (bf16)y;
    } else {
      const int cc = c - 2 * DM;
      vt[((size_t)(b * HEADS + (cc >> 6)) * DHEAD + (cc & 63)) * SEQ + l] = (bf16)y;
    }
  }
  for (int j = tid; j < DBI; j += 256) {
    const int c1 = 3 * DM + j, c2 = 3 * DM + DBI + j;
    const float y1 = ((float)hrow[c1] - mean) * rstd * g[c1] + be[c1];
    const float y2 = ((float)hrow[c2] - mean) * rstd * g[c2] + be[c2];
    op[(size_t)row * DCAT + DM + j] = (bf16)(y1 * y2);
  }
}

// ---------------------------------------------------------------------------
// Flash attention, 1 wave per 16-query tile; all fragments are b128 loads.
// S^T = K * Q^T (softmax: in-lane + one xor-16), O^T = V^T * P^T.
// ---------------------------------------------------------------------------
__global__ void __launch_bounds__(128) flash_attn(const bf16* __restrict__ qg,
                                                  const bf16* __restrict__ kg,
                                                  const bf16* __restrict__ vt,
                                                  const float* __restrict__ slopes,
                                                  bf16* __restrict__ op) {
  const int wid = blockIdx.x * 4 + (threadIdx.x >> 5);
  const int lane = threadIdx.x & 31, ln = lane & 15, hf = lane >> 4;
  const int qt = wid & (SEQ / 16 - 1);
  const int bh = wid >> 7;                  // SEQ/16 = 128 tiles per (b,h)
  const int h = bh & (HEADS - 1);
  const int b = bh >> 4;
  const int qbase = qt * 16;
  const int qglob = qbase + ln;
  const float slope = slopes[h];
  const size_t hbase = (size_t)bh * SEQ * DHEAD;

  // Q^T B-fragments: lane = query col; K rows 16*hf..16*hf+15 are contiguous.
  Frag16 bq[2];
#pragma unroll
  for (int c = 0; c < 2; ++c)
    bq[c].v = *(const v16bf*)(qg + hbase + (size_t)qglob * DHEAD + 32 * c + 16 * hf);

  v8f o[4] = {};              // O^T: rows = head dim (4 chunks of 16), cols = queries
  float mi = -1e30f, li = 0.f;

  for (int jb = 0; jb < qbase + 16; jb += 32) {
    // K tile A-fragments: d range {16hf..16hf+15} is contiguous.
    Frag16 ak[2][2];
#pragma unroll
    for (int s = 0; s < 2; ++s) {
      const bf16* krow = kg + hbase + (size_t)(jb + 16 * s + ln) * DHEAD;
#pragma unroll
      for (int c = 0; c < 2; ++c) {
        const bf16* kp = krow + 32 * c + 16 * hf;
        ak[s][c].h[0] = *(const bf16x8*)kp;
        ak[s][c].h[1] = *(const bf16x8*)(kp + 8);
      }
    }
    v8f st0 = {}, st1 = {};
    st0 = __builtin_amdgcn_wmma_f32_16x16x32_bf16(false, ak[0][0].v, false, bq[0].v, (short)0, st0, false, false);
    st0 = __builtin_amdgcn_wmma_f32_16x16x32_bf16(false, ak[0][1].v, false, bq[1].v, (short)0, st0, false, false);
    st1 = __builtin_amdgcn_wmma_f32_16x16x32_bf16(false, ak[1][0].v, false, bq[0].v, (short)0, st1, false, false);
    st1 = __builtin_amdgcn_wmma_f32_16x16x32_bf16(false, ak[1][1].v, false, bq[1].v, (short)0, st1, false, false);

    // logits + ALiBi + causal mask; lane holds keys {jb + r + 8*hf} and +16
    float p0[8], p1[8];
    float tmax = -1e30f;
#pragma unroll
    for (int r = 0; r < 8; ++r) {
      const int j0 = jb + r + 8 * hf;
      float v0 = st0[r] * 0.125f + slope * (float)j0;
      if (j0 > qglob) v0 = -1e30f;
      p0[r] = v0; tmax = fmaxf(tmax, v0);
      const int j1 = j0 + 16;
      float v1 = st1[r] * 0.125f + slope * (float)j1;
      if (j1 > qglob) v1 = -1e30f;
      p1[r] = v1; tmax = fmaxf(tmax, v1);
    }
    tmax = fmaxf(tmax, __shfl_xor(tmax, 16, 32));
    const float mnew = fmaxf(mi, tmax);
    const float corr = __expf(mi - mnew);
    float psum = 0.f;
#pragma unroll
    for (int r = 0; r < 8; ++r) {
      p0[r] = __expf(p0[r] - mnew);
      p1[r] = __expf(p1[r] - mnew);
      psum += p0[r] + p1[r];
    }
    psum += __shfl_xor(psum, 16, 32);
    li = li * corr + psum;
    mi = mnew;
#pragma unroll
    for (int t = 0; t < 4; ++t)
#pragma unroll
      for (int r = 0; r < 8; ++r) o[t][r] *= corr;

    // Repack P^T into B-fragment layout: lane half hf needs key tile s=hf,
    // local indices 0..15; indices 8..15 live in the partner lane (xor 16).
    float lowv[8], highv[8];
#pragma unroll
    for (int r = 0; r < 8; ++r) {
      const float q0 = __shfl_xor(p0[r], 16, 32);
      const float q1 = __shfl_xor(p1[r], 16, 32);
      lowv[r]  = hf ? q1 : p0[r];
      highv[r] = hf ? p1[r] : q0;
    }
    Frag16 bp;
#pragma unroll
    for (int v = 0; v < 8; ++v) {
      const int i = 2 * v;
      union { bf16 hh[2]; unsigned uu; } pk;
      pk.hh[0] = (bf16)((i < 8) ? lowv[i]     : highv[i - 8]);
      pk.hh[1] = (bf16)((i < 8) ? lowv[i + 1] : highv[i - 7]);
      bp.u[v] = pk.uu;
    }

    // O^T += V^T * P^T : A = V^T d-chunks from pre-transposed vt -> b128 loads
#pragma unroll
    for (int t = 0; t < 4; ++t) {
      Frag16 av;
      const bf16* vp = vt + ((size_t)bh * DHEAD + 16 * t + ln) * SEQ + jb + 16 * hf;
      av.h[0] = *(const bf16x8*)vp;
      av.h[1] = *(const bf16x8*)(vp + 8);
      o[t] = __builtin_amdgcn_wmma_f32_16x16x32_bf16(
          false, av.v, false, bp.v, (short)0, o[t], false, false);
    }
  }

  // Store O (lane = query ln; d = 16t + 8*hf + r contiguous) into concat buffer
  const float inv = 1.f / li;
  bf16* orow = op + (size_t)(b * SEQ + qglob) * DCAT + h * DHEAD + hf * 8;
#pragma unroll
  for (int t = 0; t < 4; ++t) {
    bf16x8 pk;
#pragma unroll
    for (int r = 0; r < 8; ++r) pk[r] = (bf16)(o[t][r] * inv);
    *(bf16x8*)(orow + 16 * t) = pk;
  }
}

// ---------------------------------------------------------------------------
// Final LayerNorm (f32 in, f32 out)
// ---------------------------------------------------------------------------
__global__ void __launch_bounds__(256) out_ln_kernel(const float* __restrict__ y,
                                                     const float* __restrict__ g,
                                                     const float* __restrict__ be,
                                                     float* __restrict__ out) {
  __shared__ float rs[256], rq[256];
  const int row = blockIdx.x;
  const int tid = threadIdx.x;
  const float* yr = y + (size_t)row * DM;
  float v[4], s = 0.f, q = 0.f;
#pragma unroll
  for (int i = 0; i < 4; ++i) {
    const int c = i * 256 + tid;
    v[i] = yr[c]; s += v[i]; q += v[i] * v[i];
  }
  rs[tid] = s; rq[tid] = q; __syncthreads();
  for (int off = 128; off; off >>= 1) {
    if (tid < off) { rs[tid] += rs[tid + off]; rq[tid] += rq[tid + off]; }
    __syncthreads();
  }
  const float mean = rs[0] * (1.f / DM);
  const float var  = rq[0] * (1.f / DM) - mean * mean;
  const float rstd = rsqrtf(var + 1e-5f);
#pragma unroll
  for (int i = 0; i < 4; ++i) {
    const int c = i * 256 + tid;
    out[(size_t)row * DM + c] = (v[i] - mean) * rstd * g[c] + be[c];
  }
}

// ---------------------------------------------------------------------------
extern "C" void kernel_launch(void* const* d_in, const int* in_sizes, int n_in,
                              void* d_out, int out_size, void* d_ws, size_t ws_size,
                              hipStream_t stream) {
  const float* x      = (const float*)d_in[0];
  const float* in_g   = (const float*)d_in[1];
  const float* in_b   = (const float*)d_in[2];
  const float* mid_g  = (const float*)d_in[3];
  const float* mid_b  = (const float*)d_in[4];
  const float* out_g  = (const float*)d_in[5];
  const float* out_b  = (const float*)d_in[6];
  const float* w_in   = (const float*)d_in[7];
  const float* w_out  = (const float*)d_in[8];
  const float* slopes = (const float*)d_in[9];

  char* p = (char*)d_ws;
  auto alloc = [&](size_t bytes) {
    char* r = p;
    p += (bytes + 255) & ~(size_t)255;
    return r;
  };
  bf16*  x16    = (bf16*)alloc((size_t)TOKENS * DM * 2);
  bf16*  w_in_pk = (bf16*)alloc((size_t)DM * DPROJ * 2);
  bf16*  w_out_pk= (bf16*)alloc((size_t)DCAT * DM * 2);
  bf16*  h16    = (bf16*)alloc((size_t)TOKENS * DPROJ * 2);
  bf16*  qb     = (bf16*)alloc((size_t)TOKENS * DM * 2);
  bf16*  kb     = (bf16*)alloc((size_t)TOKENS * DM * 2);
  bf16*  vt     = (bf16*)alloc((size_t)TOKENS * DM * 2);
  bf16*  op     = (bf16*)alloc((size_t)TOKENS * DCAT * 2);
  float* ybuf   = (float*)alloc((size_t)TOKENS * DM * 4);

  pack_weights<<<2048, 256, 0, stream>>>(w_in, w_in_pk, DM, DPROJ);
  pack_weights<<<1024, 256, 0, stream>>>(w_out, w_out_pk, DCAT, DM);
  ln_shift_kernel<<<TOKENS, 256, 0, stream>>>(x, in_g, in_b, x16);
  // GEMM1: waves = (4096/32) * (7168/64) = 14336 -> 1792 blocks of 8 waves
  gemm_wmma<true><<<(TOKENS / 32) * (DPROJ / 64) / 8, 256, 0, stream>>>(
      x16, w_in_pk, h16, TOKENS, DPROJ, DM);
  midln_split<<<TOKENS, 256, 0, stream>>>(h16, mid_g, mid_b, qb, kb, vt, op);
  flash_attn<<<(BATCH * HEADS * (SEQ / 16)) / 4, 128, 0, stream>>>(
      qb, kb, vt, slopes, op);
  // GEMM2: waves = (4096/32) * (1024/64) = 2048 -> 256 blocks
  gemm_wmma<false><<<(TOKENS / 32) * (DM / 64) / 8, 256, 0, stream>>>(
      op, w_out_pk, ybuf, TOKENS, DM, DCAT);
  out_ln_kernel<<<TOKENS, 256, 0, stream>>>(ybuf, out_g, out_b, (float*)d_out);
}
